// Asymmetric_CCAttention_21114059227554
// MI455X (gfx1250) — matrix operand order repaired and checked by reading, compiled, not run on hardware
//
#include <hip/hip_runtime.h>
#include <hip/hip_bf16.h>

typedef __attribute__((ext_vector_type(16))) __bf16       v16bf;
typedef __attribute__((ext_vector_type(8)))  float        v8f;
typedef __attribute__((ext_vector_type(4)))  float        f32x4;
typedef __attribute__((ext_vector_type(4)))  unsigned int u32x4;
typedef __attribute__((ext_vector_type(4)))  unsigned int v4u;
typedef __attribute__((ext_vector_type(8)))  int          v8i;
typedef __attribute__((ext_vector_type(4)))  int          v4i;

#define BQ   8
#define CCH  512
#define CQK  64
#define HH   128
#define WW   128
#define HWP  16384   // H*W

#if __has_builtin(__builtin_amdgcn_tensor_load_to_lds)
#define USE_TDM 1
#else
#define USE_TDM 0
#endif

union FragU { v16bf v; u32x4 q[2]; };

__device__ __forceinline__ unsigned short f2b(float x) {
  __bf16 h = (__bf16)x;
  unsigned short u;
  __builtin_memcpy(&u, &h, 2);
  return u;
}

__device__ __forceinline__ v8f wmma_bf16(v16bf a, v16bf b, v8f c) {
  return __builtin_amdgcn_wmma_f32_16x16x32_bf16(false, a, false, b, (short)0, c, false, false);
}

// ---- LDS fragment loaders (row-major bf16 panels, strides in halfwords) ----
// A (16x32, MxK): lanes 0-15 M=lane K=0-7/16-23, lanes 16-31 M=lane-16 K=8-15/24-31
__device__ __forceinline__ v16bf lds_frag_a(const unsigned int* lds, int ldaH,
                                            int m0, int k0, int lane) {
  int m  = m0 + (lane & 15);
  int kh = (lane >> 4) * 4;              // dword offset (8 halfwords)
  int base = ((m * ldaH + k0) >> 1) + kh;
  FragU u;
  u.q[0] = *(const u32x4*)(lds + base);
  u.q[1] = *(const u32x4*)(lds + base + 8);
  return u.v;
}
// B (32x16, KxN): lanes 0-15 N=lane K=0-7/16-23, lanes 16-31 N=lane-16 K=8-15/24-31
__device__ __forceinline__ v16bf lds_frag_b(const unsigned int* lds, int ldbH,
                                            int k0, int n0, int lane) {
  const __bf16* p = (const __bf16*)lds + (k0 + (lane >> 4) * 8) * ldbH + n0 + (lane & 15);
  FragU u;
#pragma unroll
  for (int i = 0; i < 8; ++i) {
    u.v[i]     = p[i * ldbH];
    u.v[8 + i] = p[(16 + i) * ldbH];
  }
  return u.v;
}

// ---- TDM: 1-D tile (4-byte elements) global -> LDS, per ISA 08_async_tensor ----
__device__ __forceinline__ void tdm_load_1d(unsigned int ldsByteOff, const void* gptr,
                                            unsigned int ndw) {
#if USE_TDM
  unsigned long long ga = (unsigned long long)(size_t)gptr;
  v4u g0 = { 1u,                                   // count=1 (valid user descriptor)
             ldsByteOff,                           // lds_addr
             (unsigned int)(ga & 0xFFFFFFFFull),   // global_addr lo
             (unsigned int)((ga >> 32) & 0x1FFFFFFull) | (2u << 30) }; // hi | type=2
  v8i g1 = { (int)(2u << 16),                      // data_size = 2 (4B), no mask/pad
             (int)((ndw & 0xFFFFu) << 16),         // tensor_dim0[15:0]
             (int)((ndw >> 16) | (1u << 16)),      // tensor_dim0[31:16] | tensor_dim1=1
             (int)((ndw & 0xFFFFu) << 16),         // tile_dim0 = ndw
             1,                                    // tile_dim1 = 1, tile_dim2 = 0
             (int)ndw,                             // tensor_dim0_stride lo
             0, 0 };
  v4i z4 = { 0, 0, 0, 0 };
  v8i z8 = { 0, 0, 0, 0, 0, 0, 0, 0 };
  __builtin_amdgcn_tensor_load_to_lds(g0, g1, z4, z4, z8, 0);
#else
  (void)ldsByteOff; (void)gptr; (void)ndw;
#endif
}

__device__ __forceinline__ void load_panel(unsigned int* ldsDst, const void* g,
                                           unsigned int ndw, int tid, int nthr) {
#if USE_TDM
  if (tid < 32) tdm_load_1d((unsigned int)(size_t)ldsDst, g, ndw);
  (void)nthr;
#else
  const u32x4* s = (const u32x4*)g;
  u32x4* d = (u32x4*)ldsDst;
  for (unsigned int i = tid; i < (ndw >> 2); i += nthr) d[i] = s[i];
#endif
}

__device__ __forceinline__ void panels_ready(int tid) {
#if USE_TDM
  if (tid < 32) __builtin_amdgcn_s_wait_tensorcnt(0);
#endif
  __syncthreads();
}

// ---- global f32 fragment loaders with inline cvt to bf16 ----
__device__ __forceinline__ v16bf gld_frag_a(const float* __restrict__ Wm, int ldw,
                                            int mBase, int k0, int lane) {
  const float* r = Wm + (size_t)(mBase + (lane & 15)) * ldw + k0 + ((lane >> 4) * 8);
  f32x4 a0 = *(const f32x4*)(r);
  f32x4 a1 = *(const f32x4*)(r + 4);
  f32x4 a2 = *(const f32x4*)(r + 16);
  f32x4 a3 = *(const f32x4*)(r + 20);
  FragU u;
#pragma unroll
  for (int i = 0; i < 4; ++i) {
    u.v[i]      = (__bf16)a0[i];
    u.v[4 + i]  = (__bf16)a1[i];
    u.v[8 + i]  = (__bf16)a2[i];
    u.v[12 + i] = (__bf16)a3[i];
  }
  return u.v;
}
__device__ __forceinline__ v16bf gld_frag_b(const float* __restrict__ X, int ldx,
                                            int k0, int n0, int lane) {
  const float* r = X + (size_t)(k0 + (lane >> 4) * 8) * ldx + n0 + (lane & 15);
  FragU u;
#pragma unroll
  for (int i = 0; i < 8; ++i) {
    u.v[i]     = (__bf16)r[(size_t)i * ldx];
    u.v[8 + i] = (__bf16)r[(size_t)(16 + i) * ldx];
  }
  return u.v;
}

// ============ 1) 1x1-conv projections: out[b][pixel][oc] (pixel-major bf16) ============
__global__ __launch_bounds__(32) void proj_kernel(const float* __restrict__ x,
                                                  const float* __restrict__ Wm,
                                                  const float* __restrict__ bias,
                                                  unsigned short* __restrict__ outp,
                                                  int Mtot) {
  __shared__ __align__(16) unsigned short tile[32 * 40];
  const int lane = threadIdx.x;
  const int p0 = blockIdx.x * 32, m0 = blockIdx.y * 32, b = blockIdx.z;
  const float* X = x + (size_t)b * CCH * HWP;
  v8f acc[2][2] = {};
  for (int k0 = 0; k0 < CCH; k0 += 32) {
    v16bf a0 = gld_frag_a(Wm, CCH, m0, k0, lane);
    v16bf a1 = gld_frag_a(Wm, CCH, m0 + 16, k0, lane);
    v16bf b0 = gld_frag_b(X, HWP, k0, p0, lane);
    v16bf b1 = gld_frag_b(X, HWP, k0, p0 + 16, lane);
    acc[0][0] = wmma_bf16(a0, b0, acc[0][0]);
    acc[0][1] = wmma_bf16(a0, b1, acc[0][1]);
    acc[1][0] = wmma_bf16(a1, b0, acc[1][0]);
    acc[1][1] = wmma_bf16(a1, b1, acc[1][1]);
  }
  const int n_ = lane & 15, hi = lane >> 4;
#pragma unroll
  for (int mt = 0; mt < 2; ++mt)
#pragma unroll
    for (int nt = 0; nt < 2; ++nt)
#pragma unroll
      for (int r = 0; r < 8; ++r) {
        int m = mt * 16 + r + 8 * hi;
        int nn = nt * 16 + n_;
        tile[nn * 40 + m] = f2b(acc[mt][nt][r] + bias[m0 + m]);
      }
  __syncthreads();
  unsigned short* dst = outp + ((size_t)b * HWP + p0 + lane) * Mtot + m0;
#pragma unroll
  for (int j = 0; j < 4; ++j) {
    u32x4 t;
#pragma unroll
    for (int l = 0; l < 4; ++l) {
      unsigned int lo = tile[lane * 40 + j * 8 + 2 * l];
      unsigned int hh = tile[lane * 40 + j * 8 + 2 * l + 1];
      t[l] = lo | (hh << 16);
    }
    *(u32x4*)(dst + j * 8) = t;
  }
}

// ============ 2a) H<->W transpose of pixel-major vectors (per (b, w) column) ============
__global__ __launch_bounds__(256) void hwt_kernel(const unsigned short* __restrict__ in,
                                                  unsigned short* __restrict__ outp,
                                                  int nqShift) {
  const int o = blockIdx.x, b = blockIdx.y;
  const int nq = 1 << nqShift;  // uint4 per pixel-vector
  const u32x4* src = (const u32x4*)in;
  u32x4* dst = (u32x4*)outp + ((size_t)b * WW + o) * HH * nq;
  const int total = HH * nq;
  for (int i = threadIdx.x; i < total; i += 256) {
    int h = i >> nqShift, j = i & (nq - 1);
    dst[i] = src[((size_t)b * HWP + h * WW + o) * nq + j];
  }
}

// ==== 2b) pixel-major k (64 ch) -> channel-major plane out[b][o][c][y] via LDS transpose ====
__global__ __launch_bounds__(256) void cm_kernel(const unsigned short* __restrict__ in,
                                                 unsigned short* __restrict__ outp,
                                                 int oStride, int yStride) {
  __shared__ __align__(16) unsigned int lds[128 * 33];  // rows of 66 halfwords (pad)
  const int o = blockIdx.x, b = blockIdx.y, tid = threadIdx.x;
  for (int i = tid; i < 128 * 8; i += 256) {
    int y = i >> 3, j = i & 7;
    size_t p = (size_t)b * HWP + (size_t)o * oStride + (size_t)y * yStride;
    u32x4 v = *((const u32x4*)in + p * 8 + j);
    int bd = y * 33 + j * 4;
    lds[bd] = v[0]; lds[bd + 1] = v[1]; lds[bd + 2] = v[2]; lds[bd + 3] = v[3];
  }
  __syncthreads();
  unsigned int* dstd = (unsigned int*)outp + ((size_t)b * 128 + o) * 64 * 64;
  const unsigned short* ldsh = (const unsigned short*)lds;
  for (int i = tid; i < 64 * 64; i += 256) {
    int c = i >> 6, j = i & 63;
    unsigned int lo = ldsh[(2 * j) * 66 + c];
    unsigned int hh = ldsh[(2 * j + 1) * 66 + c];
    dstd[c * 64 + j] = lo | (hh << 16);
  }
}

// ============ 3) attention logits: 128x128 = (128x64) x (64x128) per plane ============
__global__ __launch_bounds__(256) void att_kernel(const unsigned short* __restrict__ qA,
                                                  const unsigned short* __restrict__ kB,
                                                  float* __restrict__ logits,
                                                  const int diag) {
  __shared__ __align__(16) unsigned int lds[8192];  // A:[0,4096) B:[4096,8192) dwords
  const int o = blockIdx.x, b = blockIdx.y, tid = threadIdx.x;
  const size_t plane = (size_t)b * 128 + o;
  load_panel(lds, qA + plane * (128 * 64), 4096, tid, 256);
  load_panel(lds + 4096, kB + plane * (64 * 128), 4096, tid, 256);
  panels_ready(tid);
  const int lane = tid & 31, m0 = (tid >> 5) * 16;
  v8f acc[8] = {};
  for (int ks = 0; ks < 64; ks += 32) {
    v16bf a = lds_frag_a(lds, 64, m0, ks, lane);
#pragma unroll
    for (int j = 0; j < 8; ++j) {
      v16bf bb = lds_frag_b(lds + 4096, 128, ks, j * 16, lane);
      acc[j] = wmma_bf16(a, bb, acc[j]);
    }
  }
  const int n_ = lane & 15, hi = lane >> 4;
  float* lrow = logits + plane * (128 * 128);
#pragma unroll
  for (int j = 0; j < 8; ++j)
#pragma unroll
    for (int r = 0; r < 8; ++r) {
      int h = m0 + r + 8 * hi;
      int hp = j * 16 + n_;
      float v = acc[j][r];
      if (diag && h == hp) v = -__builtin_inff();
      lrow[h * 128 + hp] = v;
    }
}

// ============ 4) joint softmax over 256 logits per pixel, probs -> bf16 ============
__global__ __launch_bounds__(256) void softmax_kernel(const float* __restrict__ lH,
                                                      const float* __restrict__ lW,
                                                      unsigned short* __restrict__ pH,
                                                      unsigned short* __restrict__ pW) {
  const int pix = blockIdx.x * 256 + threadIdx.x;
  const int b = pix >> 14, rem = pix & 16383, h = rem >> 7, w = rem & 127;
  const size_t rH = ((size_t)b * 128 + w) * 128 + h;
  const size_t rW = ((size_t)b * 128 + h) * 128 + w;
  const f32x4* hv = (const f32x4*)(lH + rH * 128);
  const f32x4* wv = (const f32x4*)(lW + rW * 128);
  float m = -__builtin_inff();
  for (int i = 0; i < 32; ++i) {
    f32x4 a = hv[i], c = wv[i];
#pragma unroll
    for (int l = 0; l < 4; ++l) { m = fmaxf(m, a[l]); m = fmaxf(m, c[l]); }
  }
  float s = 0.f;
  for (int i = 0; i < 32; ++i) {
    f32x4 a = hv[i], c = wv[i];
#pragma unroll
    for (int l = 0; l < 4; ++l) { s += __expf(a[l] - m); s += __expf(c[l] - m); }
  }
  const float inv = 1.0f / s;
  unsigned int* oHd = (unsigned int*)(pH + rH * 128);
  unsigned int* oWd = (unsigned int*)(pW + rW * 128);
  for (int i = 0; i < 32; ++i) {
    f32x4 a = hv[i], c = wv[i];
    oHd[2 * i]     = f2b(__expf(a[0] - m) * inv) | ((unsigned)f2b(__expf(a[1] - m) * inv) << 16);
    oHd[2 * i + 1] = f2b(__expf(a[2] - m) * inv) | ((unsigned)f2b(__expf(a[3] - m) * inv) << 16);
    oWd[2 * i]     = f2b(__expf(c[0] - m) * inv) | ((unsigned)f2b(__expf(c[1] - m) * inv) << 16);
    oWd[2 * i + 1] = f2b(__expf(c[2] - m) * inv) | ((unsigned)f2b(__expf(c[3] - m) * inv) << 16);
  }
}

// ===== 5) aggregation: (128x128 probs) x (128x512 v) per plane; row pass fuses combine =====
__global__ __launch_bounds__(512) void agg_kernel(const unsigned short* __restrict__ probs,
                                                  const unsigned short* __restrict__ vmat,
                                                  float* __restrict__ oH,
                                                  const float* __restrict__ xq,
                                                  float* __restrict__ outp,
                                                  const float* __restrict__ gamma,
                                                  const int doCombine) {
  __shared__ __align__(16) unsigned int lds[40960];  // 160KB: A:[0,8192) B:[8192,40960)
  const int o = blockIdx.x, b = blockIdx.y, tid = threadIdx.x;
  const size_t plane = (size_t)b * 128 + o;
  load_panel(lds, probs + plane * 16384, 8192, tid, 512);
  load_panel(lds + 8192, vmat + plane * 65536, 32768, tid, 512);
  panels_ready(tid);
  const int lane = tid & 31, wave = tid >> 5;
  const int m0 = (wave & 3) * 32;
  const int ch = wave >> 2;       // c-quarter (128 channels)
  const int nb = ch * 128;
  v8f acc[2][8] = {};
  for (int ks = 0; ks < 128; ks += 32) {
    v16bf a0 = lds_frag_a(lds, 128, m0, ks, lane);
    v16bf a1 = lds_frag_a(lds, 128, m0 + 16, ks, lane);
#pragma unroll
    for (int j = 0; j < 8; ++j) {
      v16bf bb = lds_frag_b(lds + 8192, 512, ks, nb + j * 16, lane);
      acc[0][j] = wmma_bf16(a0, bb, acc[0][j]);
      acc[1][j] = wmma_bf16(a1, bb, acc[1][j]);
    }
  }
  const int n_ = lane & 15, hi = lane >> 4;
  if (!doCombine) {
    // o_H[b][h][w=o][c], lanes -> contiguous c (64B per half-wave store)
#pragma unroll
    for (int t = 0; t < 2; ++t)
#pragma unroll
      for (int j = 0; j < 8; ++j)
#pragma unroll
        for (int r = 0; r < 8; ++r) {
          int h = m0 + t * 16 + r + 8 * hi;
          int c = nb + j * 16 + n_;
          oH[((size_t)((b * 128 + h) * 128 + o)) * 512 + c] = acc[t][j][r];
        }
  } else {
    const float g = gamma[0];
    float* ldsf = (float*)lds;                  // staging [w][cc], stride 257 (pad)
    const float* oHp = oH + plane * 65536;      // o_H plane for this (b, h=o)
    for (int half = 0; half < 2; ++half) {
      __syncthreads();                          // frags / previous combine done
      if ((ch >> 1) == half) {
#pragma unroll
        for (int t = 0; t < 2; ++t)
#pragma unroll
          for (int j = 0; j < 8; ++j)
#pragma unroll
            for (int r = 0; r < 8; ++r) {
              int w = m0 + t * 16 + r + 8 * hi;
              int c = nb + j * 16 + n_;
              float v = acc[t][j][r] + oHp[w * 512 + c];
              ldsf[w * 257 + (c - half * 256)] = v;
            }
      }
      __syncthreads();
      for (int cc0 = 0; cc0 < 256; cc0 += 4) {
        int cc = cc0 + (tid >> 7);
        int w = tid & 127;
        int c = half * 256 + cc;
        float v = ldsf[w * 257 + cc];
        size_t idx = (((size_t)b * 512 + c) * 128 + o) * 128 + w;
        outp[idx] = g * v + xq[idx];
      }
    }
  }
}

extern "C" void kernel_launch(void* const* d_in, const int* in_sizes, int n_in,
                              void* d_out, int out_size, void* d_ws, size_t ws_size,
                              hipStream_t stream) {
  (void)in_sizes; (void)n_in; (void)out_size; (void)ws_size;
  const float* x_q   = (const float*)d_in[0];
  const float* x_kv  = (const float*)d_in[1];
  const float* Wq    = (const float*)d_in[2];
  const float* bq    = (const float*)d_in[3];
  const float* Wk    = (const float*)d_in[4];
  const float* bk    = (const float*)d_in[5];
  const float* Wv    = (const float*)d_in[6];
  const float* bv    = (const float*)d_in[7];
  const float* gamma = (const float*)d_in[8];
  float* out = (float*)d_out;
  char* ws = (char*)d_ws;

  constexpr size_t NPIX = (size_t)BQ * HWP;           // 131072
  constexpr size_t SZ_Q = NPIX * CQK * 2;             // 16 MB (bf16)
  constexpr size_t SZ_V = NPIX * CCH * 2;             // 128 MB (bf16)
  constexpr size_t SZ_L = NPIX * 128 * 4;             // 64 MB (f32)
  constexpr size_t SZ_P = NPIX * 128 * 2;             // 32 MB (bf16)

  unsigned short* q_pix = (unsigned short*)(ws);                       // [b][h][w][c64]
  unsigned short* k_pix = (unsigned short*)(ws + SZ_Q);                // [b][h][w][c64]
  unsigned short* qAH   = (unsigned short*)(ws + 2 * SZ_Q);            // [b][w][h][c64]
  unsigned short* kBH   = (unsigned short*)(ws + 3 * SZ_Q);            // [b][w][c][h]
  unsigned short* kBWr  = (unsigned short*)(ws + 4 * SZ_Q);            // [b][h][c][w]
  unsigned short* v_pix = (unsigned short*)(ws + 5 * SZ_Q);            // [b][h][w][c512]
  unsigned short* vH    = (unsigned short*)(ws + 5 * SZ_Q + SZ_V);     // [b][w][h][c512]
  float* logH = (float*)(ws + 5 * SZ_Q + 2 * SZ_V);                    // [b][w][h][h']
  float* logW = (float*)(ws + 5 * SZ_Q + 2 * SZ_V + SZ_L);             // [b][h][w][w']
  unsigned short* pH = (unsigned short*)(ws + 5 * SZ_Q + 2 * SZ_V + 2 * SZ_L);
  unsigned short* pW = (unsigned short*)(ws + 5 * SZ_Q + 2 * SZ_V + 2 * SZ_L + SZ_P);
  float* oHbuf = (float*)(ws + 5 * SZ_Q + 2 * SZ_V + 2 * SZ_L + 2 * SZ_P);  // [b][h][w][c] f32

  // projections (q,k,v) -> pixel-major bf16
  proj_kernel<<<dim3(512, 2, 8),  32, 0, stream>>>(x_q,  Wq, bq, q_pix, CQK);
  proj_kernel<<<dim3(512, 2, 8),  32, 0, stream>>>(x_kv, Wk, bk, k_pix, CQK);
  proj_kernel<<<dim3(512, 16, 8), 32, 0, stream>>>(x_kv, Wv, bv, v_pix, CCH);
  // repacks
  hwt_kernel<<<dim3(128, 8), 256, 0, stream>>>(q_pix, qAH, 3);
  hwt_kernel<<<dim3(128, 8), 256, 0, stream>>>(v_pix, vH, 6);
  cm_kernel<<<dim3(128, 8), 256, 0, stream>>>(k_pix, kBH, 1, 128);    // per-column panel
  cm_kernel<<<dim3(128, 8), 256, 0, stream>>>(k_pix, kBWr, 128, 1);   // per-row panel
  // logits (e_H with diagonal mask, e_W)
  att_kernel<<<dim3(128, 8), 256, 0, stream>>>(qAH,   kBH,  logH, 1);
  att_kernel<<<dim3(128, 8), 256, 0, stream>>>(q_pix, kBWr, logW, 0);
  // joint softmax over 256
  softmax_kernel<<<512, 256, 0, stream>>>(logH, logW, pH, pW);
  // aggregation: column pass writes o_H, row pass fuses o_W + gamma + residual
  agg_kernel<<<dim3(128, 8), 512, 0, stream>>>(pH, vH,    oHbuf, nullptr, nullptr, nullptr, 0);
  agg_kernel<<<dim3(128, 8), 512, 0, stream>>>(pW, v_pix, oHbuf, x_q,     out,     gamma,  1);
}
